// DeGTAConv_69947837383032
// MI455X (gfx1250) — matrix operand order repaired and checked by reading, not compiled, not run on hardware
//
#include <hip/hip_runtime.h>
#include <math.h>

// ---------------------------------------------------------------------------
// DeGTAConv forward for MI455X (gfx1250, wave32, WMMA bf16 16x16x32)
// ---------------------------------------------------------------------------

#define N_NODES 4096
#define CH      256
#define CH2     512
#define N_EDGES 131072

typedef __bf16 bf16_t;
typedef bf16_t v16bf __attribute__((ext_vector_type(16)));
typedef float  v8f   __attribute__((ext_vector_type(8)));

__device__ __forceinline__ unsigned short f32_to_bf16(float f) {
  unsigned u = __float_as_uint(f);
  u += 0x7FFFu + ((u >> 16) & 1u);          // round-to-nearest-even
  return (unsigned short)(u >> 16);
}

// two bf16 packed into one dword (lo in [15:0], hi in [31:16])
__device__ __forceinline__ unsigned pack_bf16x2(float lo, float hi) {
  return (unsigned)f32_to_bf16(lo) | ((unsigned)f32_to_bf16(hi) << 16);
}

__device__ __forceinline__ v16bf load_frag(const unsigned short* p0,
                                           const unsigned short* p1) {
  union { uint4 u[2]; v16bf v; } t;
  t.u[0] = *(const uint4*)p0;               // ds_load_b128
  t.u[1] = *(const uint4*)p1;
  return t.v;
}

// ---------------------------------------------------------------------------
// Tiled bf16 WMMA GEMM:  C[M,N] = alpha * A[M,K] @ op(B) (+ bias) (+= if ACCUM)
//   TRANSB=0: B is [K,N] row-major.  TRANSB=1: B is [N,K] row-major (A@B^T).
//   All of M,N multiples of 128; K multiple of 32 (true for every call here).
// Block = 256 threads = 8 waves; each wave computes a 32x64 patch
// (2x4 tiles of 16x16), one v_wmma_f32_16x16x32_bf16 per tile per K-chunk.
// LDS staging converts f32 -> bf16 with packed 64-bit DS stores.
// ---------------------------------------------------------------------------
#define BM 128
#define BN 128
#define BK 32

template<bool RELU, bool TRANSB, bool ACCUM>
__global__ __launch_bounds__(256)
void gemm_bf16_wmma(const float* __restrict__ A, const float* __restrict__ B,
                    const float* __restrict__ bias, float* __restrict__ C,
                    int M, int N, int K, float alpha)
{
  __shared__ unsigned short As[BM][BK];   // [m][k]
  __shared__ unsigned short Bs[BN][BK];   // [n][k]  (B^T tile)

  const int tid  = threadIdx.x;
  const int lane = tid & 31;
  const int wave = tid >> 5;
  const int bm = blockIdx.y * BM;
  const int bn = blockIdx.x * BN;
  const int wm = (wave & 3) * 32;         // wave row offset within tile
  const int wn = (wave >> 2) * 64;        // wave col offset within tile

  const int lrow = lane & 15;
  const int kh8  = (lane >> 4) << 3;      // A frag: K-half select {0,8}
  const int kh16 = (lane >> 4) << 4;      // unused; B frag also uses kh8

  v8f acc[2][4];
#pragma unroll
  for (int i = 0; i < 2; ++i)
#pragma unroll
    for (int j = 0; j < 4; ++j)
#pragma unroll
      for (int r = 0; r < 8; ++r) acc[i][j][r] = 0.0f;

  for (int k0 = 0; k0 < K; k0 += BK) {
    // ---- stage A tile (f32 -> bf16), 128x32: float4 load, ds_store_b64 ----
#pragma unroll
    for (int i = 0; i < 4; ++i) {
      int idx = tid + i * 256;
      int m   = idx >> 3;
      int kq  = (idx & 7) << 2;
      float4 v = *(const float4*)(A + (size_t)(bm + m) * K + k0 + kq);
      *(uint2*)&As[m][kq] =
          make_uint2(pack_bf16x2(v.x, v.y), pack_bf16x2(v.z, v.w));
    }
    // ---- stage B tile as B^T (n-major) ----
    if (TRANSB) {
#pragma unroll
      for (int i = 0; i < 4; ++i) {
        int idx = tid + i * 256;
        int n   = idx >> 3;
        int kq  = (idx & 7) << 2;
        float4 v = *(const float4*)(B + (size_t)(bn + n) * K + k0 + kq);
        *(uint2*)&Bs[n][kq] =
            make_uint2(pack_bf16x2(v.x, v.y), pack_bf16x2(v.z, v.w));
      }
    } else {
      // B row-major [K,N]: per thread take 4 k-consecutive elements of one
      // column (4 coalesced b32 loads, lanes sweep n), emit one ds_store_b64.
#pragma unroll
      for (int i = 0; i < 4; ++i) {
        int idx = tid + i * 256;
        int n   = idx & 127;
        int kq  = (idx >> 7) << 2;
        const float* bp = B + (size_t)(k0 + kq) * N + bn + n;
        float b0 = bp[0];
        float b1 = bp[(size_t)N];
        float b2 = bp[(size_t)2 * N];
        float b3 = bp[(size_t)3 * N];
        *(uint2*)&Bs[n][kq] =
            make_uint2(pack_bf16x2(b0, b1), pack_bf16x2(b2, b3));
      }
    }
    __syncthreads();

    // ---- fragments per documented bf16 lane layouts ----
    v16bf af[2], bfv[4];
#pragma unroll
    for (int i = 0; i < 2; ++i) {
      const unsigned short* p = &As[wm + i * 16 + lrow][kh8];
      af[i] = load_frag(p, p + 16);     // K = kh8..kh8+7, 16+kh8..16+kh8+7
    }
#pragma unroll
    for (int j = 0; j < 4; ++j) {
      const unsigned short* p = &Bs[wn + j * 16 + lrow][kh8];
      bfv[j] = load_frag(p, p + 16);    // K = kh8..kh8+7, 16+kh8..16+kh8+7
    }
#pragma unroll
    for (int i = 0; i < 2; ++i)
#pragma unroll
      for (int j = 0; j < 4; ++j)
        acc[i][j] = __builtin_amdgcn_wmma_f32_16x16x32_bf16(
            false, af[i], false, bfv[j], (short)0, acc[i][j], false, false);
    __syncthreads();
  }

  // ---- writeback: C/D layout: lane<16 -> M=r, lane>=16 -> M=8+r; N=lane&15
#pragma unroll
  for (int i = 0; i < 2; ++i) {
    int row0 = bm + wm + i * 16 + ((lane >> 4) << 3);
#pragma unroll
    for (int j = 0; j < 4; ++j) {
      int col = bn + wn + j * 16 + (lane & 15);
      float bsv = (bias != nullptr) ? bias[col] : 0.0f;
#pragma unroll
      for (int r = 0; r < 8; ++r) {
        float v = alpha * acc[i][j][r] + bsv;
        if (RELU) v = fmaxf(v, 0.0f);
        size_t o = (size_t)(row0 + r) * N + col;
        if (ACCUM) C[o] += v; else C[o] = v;
      }
    }
  }
}

// ---------------------------------------------------------------------------
// GAT branch helpers.  (x@W)@a == x@(W@a): fold to one GEMV.
// ---------------------------------------------------------------------------
__global__ void compute_u(const float* __restrict__ gw,
                          const float* __restrict__ a_src,
                          const float* __restrict__ a_dst,
                          float* __restrict__ u, int c)
{
  int i = blockIdx.x * blockDim.x + threadIdx.x;
  if (i >= c) return;
  float s0 = 0.f, s1 = 0.f;
  for (int j = 0; j < c; ++j) {
    float g = gw[(size_t)i * c + j];
    s0 += g * a_src[j];
    s1 += g * a_dst[j];
  }
  u[i] = s0; u[c + i] = s1;
}

__global__ __launch_bounds__(256)
void gat_gemv(const float* __restrict__ x, const float* __restrict__ u,
              float* __restrict__ as_, float* __restrict__ ad_, int n, int c)
{
  int wave = threadIdx.x >> 5, lane = threadIdx.x & 31;
  int row = blockIdx.x * 8 + wave;
  if (row >= n) return;
  const float* xr = x + (size_t)row * c;
  float s0 = 0.f, s1 = 0.f;
  for (int j = lane; j < c; j += 32) {
    float xv = xr[j];
    s0 += xv * u[j];
    s1 += xv * u[c + j];
  }
  for (int o = 16; o > 0; o >>= 1) {
    s0 += __shfl_down(s0, o, 32);
    s1 += __shfl_down(s1, o, 32);
  }
  if (lane == 0) { as_[row] = s0; ad_[row] = s1; }
}

__device__ __forceinline__ unsigned fkey(float f) {
  unsigned u = __float_as_uint(f);
  return (u & 0x80000000u) ? ~u : (u | 0x80000000u);
}
__device__ __forceinline__ float funkey(unsigned k) {
  unsigned u = (k & 0x80000000u) ? (k & 0x7FFFFFFFu) : ~k;
  return __uint_as_float(u);
}

__global__ void edge_pass1(const int* __restrict__ src, const int* __restrict__ dst,
                           const float* __restrict__ as_, const float* __restrict__ ad_,
                           float* __restrict__ logits, unsigned* __restrict__ maxkey, int e)
{
  int i = blockIdx.x * blockDim.x + threadIdx.x;
  if (i >= e) return;
  float l = as_[src[i]] + ad_[dst[i]];
  l = (l > 0.f) ? l : 0.2f * l;             // leaky_relu, slope 0.2
  logits[i] = l;
  atomicMax(&maxkey[dst[i]], fkey(l));
}

__global__ void edge_pass2(const int* __restrict__ dst,
                           float* __restrict__ logits,
                           const unsigned* __restrict__ maxkey,
                           float* __restrict__ sumexp, int e)
{
  int i = blockIdx.x * blockDim.x + threadIdx.x;
  if (i >= e) return;
  float m = funkey(maxkey[dst[i]]);
  float ev = __expf(logits[i] - m);
  logits[i] = ev;
  atomicAdd(&sumexp[dst[i]], ev);
}

template<bool INIT>
__global__ void edge_alpha(const int* __restrict__ dst,
                           const float* __restrict__ logits,
                           const float* __restrict__ sumexp,
                           const float* __restrict__ coef,
                           float* __restrict__ attn, int e)
{
  int i = blockIdx.x * blockDim.x + threadIdx.x;
  if (i >= e) return;
  float a = (*coef) * logits[i] / sumexp[dst[i]];
  attn[i] = INIT ? a : (attn[i] + a);
}

__global__ void edge_count(const int* __restrict__ dst, float* __restrict__ cnt, int e)
{
  int i = blockIdx.x * blockDim.x + threadIdx.x;
  if (i >= e) return;
  atomicAdd(&cnt[dst[i]], 1.0f);
}

__global__ __launch_bounds__(256)
void local_agg(const int* __restrict__ src, const int* __restrict__ dst,
               const float* __restrict__ attn, const float* __restrict__ x,
               float* __restrict__ out, int c)
{
  int e  = blockIdx.x;
  int ch = threadIdx.x;
  float w = attn[e];
  atomicAdd(&out[(size_t)dst[e] * c + ch], w * x[(size_t)src[e] * c + ch]);
}

__global__ void local_fin(float* __restrict__ out, const float* __restrict__ cnt,
                          int n, int c)
{
  size_t i = (size_t)blockIdx.x * blockDim.x + threadIdx.x;
  if (i >= (size_t)n * c) return;
  out[i] /= fmaxf(cnt[i / c], 1.0f);
}

// ---------------------------------------------------------------------------
// Global (dense attention) branch helpers
// ---------------------------------------------------------------------------
__global__ __launch_bounds__(256)
void row_softmax(float* __restrict__ buf, int n)
{
  __shared__ float red[256];
  int row = blockIdx.x, tid = threadIdx.x;
  float* r = buf + (size_t)row * n;
  float m = -3.402823466e38f;
  for (int j = tid; j < n; j += 256) m = fmaxf(m, r[j]);
  red[tid] = m; __syncthreads();
  for (int s = 128; s > 0; s >>= 1) {
    if (tid < s) red[tid] = fmaxf(red[tid], red[tid + s]);
    __syncthreads();
  }
  m = red[0]; __syncthreads();
  float s = 0.f;
  for (int j = tid; j < n; j += 256) { float ev = __expf(r[j] - m); r[j] = ev; s += ev; }
  red[tid] = s; __syncthreads();
  for (int st = 128; st > 0; st >>= 1) {
    if (tid < st) red[tid] += red[tid + st];
    __syncthreads();
  }
  float inv = 1.0f / red[0]; __syncthreads();
  for (int j = tid; j < n; j += 256) r[j] *= inv;
}

__global__ void combine_scaled(float* __restrict__ S, const float* __restrict__ T,
                               const float* __restrict__ pa, const float* __restrict__ pb,
                               size_t n)
{
  size_t i = (size_t)blockIdx.x * blockDim.x + threadIdx.x;
  if (i >= n) return;
  S[i] = (*pa) * S[i] + (*pb) * T[i];
}

__global__ void adj_zero(const int* __restrict__ src, const int* __restrict__ dst,
                         float* __restrict__ S, int n, int e)
{
  int i = blockIdx.x * blockDim.x + threadIdx.x;
  if (i >= e) return;
  S[(size_t)src[i] * n + dst[i]] = 0.0f;
}

// Per-column top-K selection + g_attn build + column normalization.
// One block (256 threads) per column j. Column cached in LDS (16 KB).
__global__ __launch_bounds__(256)
void topk_col(float* __restrict__ S, const float* __restrict__ AW,
              const float* __restrict__ pag, const float* __restrict__ pbg,
              const float* __restrict__ pcg, int n, const int* __restrict__ Kp)
{
  __shared__ float col[N_NODES];
  __shared__ float redv[256];
  __shared__ int   redi[256];
  __shared__ int   selIdx[64];
  __shared__ float selVal[64];
  __shared__ float gv[64];
  __shared__ float colsum;

  int j = blockIdx.x, tid = threadIdx.x;
  int K = *Kp;
  if (K > 64) K = 64;

  for (int i = tid; i < n; i += 256) col[i] = S[(size_t)i * n + j];
  __syncthreads();

  for (int r = 0; r < K; ++r) {
    float bv = -INFINITY; int bi = 0x7fffffff;
    for (int i = tid; i < n; i += 256) {
      float v = col[i];
      if (v > bv || (v == bv && i < bi)) { bv = v; bi = i; }
    }
    redv[tid] = bv; redi[tid] = bi; __syncthreads();
    for (int s = 128; s > 0; s >>= 1) {
      if (tid < s) {
        float ov = redv[tid + s]; int oi = redi[tid + s];
        if (ov > redv[tid] || (ov == redv[tid] && oi < redi[tid])) {
          redv[tid] = ov; redi[tid] = oi;
        }
      }
      __syncthreads();
    }
    if (tid == 0) {
      selIdx[r] = redi[0]; selVal[r] = redv[0];
      col[redi[0]] = -INFINITY;
    }
    __syncthreads();
  }

  float c1 = 0.5f * (*pag) + 0.5f * (*pbg);
  float c2 = *pcg;
  if (tid < K) {
    int i = selIdx[tid];
    gv[tid] = c1 * selVal[tid] + c2 * AW[(size_t)i * n + j];
  }
  __syncthreads();
  if (tid == 0) {
    float s = 0.f;
    for (int r = 0; r < K; ++r) s += gv[r];
    colsum = s;
  }
  __syncthreads();
  float inv = 1.0f / colsum;
  for (int i = tid; i < n; i += 256) S[(size_t)i * n + j] = 0.0f;
  __syncthreads();
  if (tid < K) S[(size_t)selIdx[tid] * n + j] = gv[tid] * inv;
}

// ---------------------------------------------------------------------------
// Fill kernels
// ---------------------------------------------------------------------------
__global__ void fill_f32(float* p, float v, size_t n) {
  size_t i = (size_t)blockIdx.x * blockDim.x + threadIdx.x;
  if (i < n) p[i] = v;
}
__global__ void fill_u32(unsigned* p, unsigned v, size_t n) {
  size_t i = (size_t)blockIdx.x * blockDim.x + threadIdx.x;
  if (i < n) p[i] = v;
}

// ---------------------------------------------------------------------------
// Host launcher
// ---------------------------------------------------------------------------
extern "C" void kernel_launch(void* const* d_in, const int* in_sizes, int n_in,
                              void* d_out, int out_size, void* d_ws, size_t ws_size,
                              hipStream_t stream)
{
  (void)in_sizes; (void)n_in; (void)out_size; (void)ws_size;
  const int Nn = N_NODES, Cc = CH, C2 = CH2, Ee = N_EDGES;

  const float* ae = (const float*)d_in[0];
  const float* pe = (const float*)d_in[1];
  const float* se = (const float*)d_in[2];
  // per-prefix parameter block bases: ae=3, pe=14, se=25
  // offsets within block: w1=0,b1=1,w2=2,b2=3,gw=4,as=5,ad=6,wq=7,bq=8,wk=9,bk=10
  auto P = [&](int base, int off) { return (const float*)d_in[base + off]; };
  const int AEB = 3, PEB = 14, SEB = 25;

  const float* a_l = (const float*)d_in[36];
  const float* b_l = (const float*)d_in[37];
  const float* c_l = (const float*)d_in[38];
  const float* a_g = (const float*)d_in[39];
  const float* b_g = (const float*)d_in[40];
  const float* c_g = (const float*)d_in[41];
  const float* w_local  = (const float*)d_in[42];
  const float* w_global = (const float*)d_in[43];
  const int* edge = (const int*)d_in[44];
  const int* src = edge;
  const int* dst = edge + Ee;
  const int* Kp  = (const int*)d_in[45];
  float* out = (float*)d_out;

  // ---- workspace carve ----
  char* ws = (char*)d_ws;
  size_t cur = 0;
  auto alloc = [&](size_t bytes) {
    void* p = ws + cur;
    cur += (bytes + 255) & ~(size_t)255;
    return p;
  };
  float* enc_ae = (float*)alloc((size_t)Nn * Cc * 4);
  float* enc_pe = (float*)alloc((size_t)Nn * Cc * 4);
  float* enc_se = (float*)alloc((size_t)Nn * Cc * 4);
  float* hidden = (float*)alloc((size_t)Nn * C2 * 4);
  float* qb     = (float*)alloc((size_t)Nn * Cc * 4);
  float* kb     = (float*)alloc((size_t)Nn * Cc * 4);
  float* out_local  = (float*)alloc((size_t)Nn * Cc * 4);
  float* out_global = (float*)alloc((size_t)Nn * Cc * 4);
  float* as_   = (float*)alloc((size_t)Nn * 4);
  float* ad_   = (float*)alloc((size_t)Nn * 4);
  float* u2    = (float*)alloc((size_t)2 * Cc * 4);
  unsigned* maxkey = (unsigned*)alloc((size_t)Nn * 4);
  float* sumexp = (float*)alloc((size_t)Nn * 4);
  float* cnt    = (float*)alloc((size_t)Nn * 4);
  float* logits = (float*)alloc((size_t)Ee * 4);
  float* attn   = (float*)alloc((size_t)Ee * 4);
  float* bufS   = (float*)alloc((size_t)Nn * Nn * 4);   // 64 MB
  float* bufT   = (float*)alloc((size_t)Nn * Nn * 4);   // 64 MB

  const int EBLK = Ee / 256;

  // ================= MLP encoders =================
  auto mlp = [&](const float* x, int base, float* enc) {
    gemm_bf16_wmma<true,  false, false><<<dim3(C2 / BN, Nn / BM), 256, 0, stream>>>(
        x, P(base, 0), P(base, 1), hidden, Nn, C2, Cc, 1.0f);
    gemm_bf16_wmma<false, false, false><<<dim3(Cc / BN, Nn / BM), 256, 0, stream>>>(
        hidden, P(base, 2), P(base, 3), enc, Nn, Cc, C2, 1.0f);
  };
  mlp(ae, AEB, enc_ae);
  mlp(pe, PEB, enc_pe);
  mlp(se, SEB, enc_se);

  // ================= local branch (fused GAT attention) =================
  auto local_branch = [&](const float* encx, int base, const float* coef, bool init) {
    compute_u<<<1, 256, 0, stream>>>(P(base, 4), P(base, 5), P(base, 6), u2, Cc);
    gat_gemv<<<Nn / 8, 256, 0, stream>>>(encx, u2, as_, ad_, Nn, Cc);
    fill_u32<<<Nn / 256, 256, 0, stream>>>(maxkey, 0u, (size_t)Nn);
    fill_f32<<<Nn / 256, 256, 0, stream>>>(sumexp, 0.0f, (size_t)Nn);
    edge_pass1<<<EBLK, 256, 0, stream>>>(src, dst, as_, ad_, logits, maxkey, Ee);
    edge_pass2<<<EBLK, 256, 0, stream>>>(dst, logits, maxkey, sumexp, Ee);
    if (init)
      edge_alpha<true ><<<EBLK, 256, 0, stream>>>(dst, logits, sumexp, coef, attn, Ee);
    else
      edge_alpha<false><<<EBLK, 256, 0, stream>>>(dst, logits, sumexp, coef, attn, Ee);
  };
  local_branch(enc_pe, PEB, a_l, true);    // a_l * pa
  local_branch(enc_se, SEB, b_l, false);   // + b_l * sa
  local_branch(enc_ae, AEB, c_l, false);   // + c_l * aa

  fill_f32<<<Nn / 256, 256, 0, stream>>>(cnt, 0.0f, (size_t)Nn);
  edge_count<<<EBLK, 256, 0, stream>>>(dst, cnt, Ee);
  fill_f32<<<(Nn * Cc) / 256, 256, 0, stream>>>(out_local, 0.0f, (size_t)Nn * Cc);
  local_agg<<<Ee, 256, 0, stream>>>(src, dst, attn, enc_ae, out_local, Cc);
  local_fin<<<(Nn * Cc) / 256, 256, 0, stream>>>(out_local, cnt, Nn, Cc);

  // ================= global branch (dense attention maps) =================
  auto mha = [&](const float* encx, int base, float* target) {
    gemm_bf16_wmma<false, false, false><<<dim3(Cc / BN, Nn / BM), 256, 0, stream>>>(
        encx, P(base, 7), P(base, 8), qb, Nn, Cc, Cc, 1.0f);
    gemm_bf16_wmma<false, false, false><<<dim3(Cc / BN, Nn / BM), 256, 0, stream>>>(
        encx, P(base, 9), P(base, 10), kb, Nn, Cc, Cc, 1.0f);
    gemm_bf16_wmma<false, true,  false><<<dim3(Nn / BN, Nn / BM), 256, 0, stream>>>(
        qb, kb, nullptr, target, Nn, Nn, Cc, 0.0625f);   // 1/sqrt(256)
    row_softmax<<<Nn, 256, 0, stream>>>(target, Nn);
  };
  mha(enc_pe, PEB, bufS);                                   // pw
  mha(enc_se, SEB, bufT);                                   // sw
  combine_scaled<<<(unsigned)(((size_t)Nn * Nn) / 256), 256, 0, stream>>>(
      bufS, bufT, a_g, b_g, (size_t)Nn * Nn);               // sample_attn
  mha(enc_ae, AEB, bufT);                                   // aw
  adj_zero<<<EBLK, 256, 0, stream>>>(src, dst, bufS, Nn, Ee);
  topk_col<<<Nn, 256, 0, stream>>>(bufS, bufT, a_g, b_g, c_g, Nn, Kp);  // -> g_attn
  gemm_bf16_wmma<false, false, false><<<dim3(Cc / BN, Nn / BM), 256, 0, stream>>>(
      bufS, enc_ae, nullptr, out_global, Nn, Cc, Nn, 1.0f);

  // ================= final projections =================
  gemm_bf16_wmma<false, false, false><<<dim3(Cc / BN, Nn / BM), 256, 0, stream>>>(
      out_local, w_local, nullptr, out, Nn, Cc, Cc, 1.0f);
  gemm_bf16_wmma<false, false, true ><<<dim3(Cc / BN, Nn / BM), 256, 0, stream>>>(
      out_global, w_global, nullptr, out, Nn, Cc, Cc, 1.0f);
}
